// ImMatchNet_79113297592428
// MI455X (gfx1250) — compile-verified, matrix operand order checked by reading, not run on hardware
//
#include <hip/hip_runtime.h>
#include <math.h>
#include <stdint.h>

typedef __attribute__((ext_vector_type(16))) _Float16 v16h;
typedef __attribute__((ext_vector_type(8)))  _Float16 v8h;
typedef __attribute__((ext_vector_type(8)))  float    v8f;
typedef unsigned int uint32x4 __attribute__((ext_vector_type(4)));
typedef int          int32x8  __attribute__((ext_vector_type(8)));
typedef int          int32x4  __attribute__((ext_vector_type(4)));

#define NPIX 1600          // 40*40 pixels per image
#define NCH  1024          // channels
#define NP   400           // 20*20 pooled matrix dim
#define POOLN (NP*NP)      // 160000 per batch
static const size_t CORR_B = (size_t)1600 * 1600; // per-batch corr elements

// ---------------------------------------------------------------------------
// 1) L2-normalize over channel dim, emit f16 in [b][pixel][channel] layout
// ---------------------------------------------------------------------------
__global__ void l2norm_f16_kernel(const float* __restrict__ in,
                                  _Float16* __restrict__ out) {
  int idx = blockIdx.x * blockDim.x + threadIdx.x;
  if (idx >= 2 * NPIX) return;
  int b = idx / NPIX, pix = idx % NPIX;
  const float* ip = in + (size_t)b * NCH * NPIX + pix;
  float s = 0.f;
  for (int c = 0; c < NCH; ++c) { float v = ip[(size_t)c * NPIX]; s += v * v; }
  float inv = 1.0f / sqrtf(s + 1e-6f);
  _Float16* op = out + ((size_t)b * NPIX + pix) * NCH;
  for (int c = 0; c < NCH; ++c)
    op[c] = (_Float16)(ip[(size_t)c * NPIX] * inv);
}

// ---------------------------------------------------------------------------
// 2) Correlation GEMM via V_WMMA_F32_16X16X32_F16.
//    Block = (batch, M-tile). The 16x1024 f16 A tile (32 KB) is DMA'd into
//    LDS once per block by the Tensor Data Mover; all 8 waves then read A
//    fragments from LDS (ds_load_b128) while streaming B from global.
//    Per-lane fragment pattern (identical for A rows / B columns):
//      lanes 0-15 : pixel = base+lane,    K halves {0..7}  and {16..23}
//      lanes16-31 : pixel = base+lane-16, K halves {8..15} and {24..31}
// ---------------------------------------------------------------------------
__device__ inline v16h load_frag_g(const _Float16* __restrict__ F, int pix,
                                   int k0, int hi) {
  const _Float16* p = F + (size_t)pix * NCH + k0 + hi * 8;
  v16h f;
  *(v8h*)&f       = *(const v8h*)p;        // K halves  k0+hi*8    .. +7
  *((v8h*)&f + 1) = *(const v8h*)(p + 16); // K halves  k0+16+hi*8 .. +7
  return f;
}

__global__ void corr_wmma_kernel(const _Float16* __restrict__ A,
                                 const _Float16* __restrict__ B,
                                 float* __restrict__ C) {
  __shared__ _Float16 Atile[16 * NCH];   // 32 KB of the 320 KB/WGP LDS
  int b    = blockIdx.x / 100;
  int tm   = blockIdx.x % 100;
  int pbase = tm * 16;
  int lane = threadIdx.x & 31;
  int wid  = threadIdx.x >> 5;
  int mrow = lane & 15;
  int hi   = lane >> 4;

  const _Float16* Ag = A + ((size_t)b * NPIX + pbase) * NCH;
  const _Float16* Bb = B + (size_t)b * NPIX * NCH;

  if (wid == 0) {
    // --- Tensor DMA descriptor (D#): 2D tensor, 1024 x 16, 2-byte elems ---
    unsigned long long ga = (unsigned long long)(uintptr_t)Ag;
    unsigned ldsa = (unsigned)(unsigned long long)(uintptr_t)&Atile[0];
    uint32x4 g0;
    g0.x = 1u;                                           // count=1, user mode
    g0.y = ldsa;                                         // lds_addr (bytes)
    g0.z = (unsigned)(ga & 0xffffffffull);               // global_addr[31:0]
    g0.w = (unsigned)((ga >> 32) & 0x1ffffffull) | (2u << 30); // [56:32]+type=2
    int32x8 g1;
    g1[0] = (int)(1u << 16);                 // data_size = 2 bytes; mask = 0
    g1[1] = (int)(((unsigned)NCH & 0xffffu) << 16);      // tensor_dim0 lo
    g1[2] = (int)(((unsigned)NCH >> 16) | (16u << 16));  // dim0 hi | dim1 lo
    g1[3] = (int)((unsigned)NCH << 16);                  // tile_dim0 = 1024
    g1[4] = 16;                                          // tile_dim1 = 16
    g1[5] = NCH;                                         // dim0_stride lo
    g1[6] = 0;                                           // stride hi bits
    g1[7] = 0;
    int32x4 gz = {};
#if defined(__clang_major__) && (__clang_major__ >= 23)
    int32x8 gz8 = {};
    __builtin_amdgcn_tensor_load_to_lds(g0, g1, gz, gz, gz8, 0);
#else
    __builtin_amdgcn_tensor_load_to_lds(g0, g1, gz, gz, 0);
#endif
    __builtin_amdgcn_s_wait_tensorcnt(0);
  }
  __syncthreads();

  // Each wave sweeps N strips of 64 columns: s = wid, wid+8, ... (< 25)
  for (int s = wid; s < 25; s += 8) {
    int qbase = s * 64;
    v8f acc[4] = {};
    for (int k0 = 0; k0 < NCH; k0 += 32) {
      // gfx1250 global_prefetch_b8 for the next K chunk of the B panel
      __builtin_prefetch(Bb + (size_t)(qbase + mrow) * NCH + k0 + 32, 0, 1);
      const _Float16* ap = &Atile[mrow * NCH + k0 + hi * 8];
      v16h a;
      *(v8h*)&a       = *(const v8h*)ap;        // ds_load_b128
      *((v8h*)&a + 1) = *(const v8h*)(ap + 16); // ds_load_b128
#pragma unroll
      for (int t = 0; t < 4; ++t) {
        v16h bf = load_frag_g(Bb, qbase + t * 16 + mrow, k0, hi);
        acc[t] = __builtin_amdgcn_wmma_f32_16x16x32_f16(
            /*neg_a=*/false, a, /*neg_b=*/false, bf,
            /*c_mod=*/(short)0, acc[t], /*reuse_a=*/false, /*reuse_b=*/false);
      }
    }
    // C/D layout: VGPR r -> row (pbase + hi*8 + r), N = qbase + t*16 + mrow
    float* cb = C + (size_t)b * CORR_B;
    int q0 = qbase + mrow;
#pragma unroll
    for (int t = 0; t < 4; ++t) {
      float* cp = cb + (size_t)(pbase + hi * 8) * 1600 + q0 + t * 16;
#pragma unroll
      for (int rr = 0; rr < 8; ++rr) cp[(size_t)rr * 1600] = acc[t][rr];
    }
  }
}

// ---------------------------------------------------------------------------
// 3) 2x2x2x2 max-pool: corr[b][1600][1600] -> pooled[b][400][400]
// ---------------------------------------------------------------------------
__global__ void maxpool4d_kernel(const float* __restrict__ C,
                                 float* __restrict__ P) {
  int idx = blockIdx.x * blockDim.x + threadIdx.x;
  if (idx >= 2 * POOLN) return;
  int b = idx / POOLN, r = idx % POOLN;
  int p = r / NP, q = r % NP;
  int pi = p / 20, pj = p % 20, pk = q / 20, pl = q % 20;
  const float* cb = C + (size_t)b * CORR_B;
  float m = -3.4e38f;
#pragma unroll
  for (int a = 0; a < 2; ++a)
#pragma unroll
    for (int bb = 0; bb < 2; ++bb)
#pragma unroll
      for (int cc = 0; cc < 2; ++cc)
#pragma unroll
        for (int dd = 0; dd < 2; ++dd) {
          int pr = (2 * pi + a) * 40 + (2 * pj + bb);
          int qc = (2 * pk + cc) * 40 + (2 * pl + dd);
          m = fmaxf(m, cb[(size_t)pr * 1600 + qc]);
        }
  P[idx] = m;
}

// ---------------------------------------------------------------------------
// 4) mutual matching helpers on the [b][400][400] matrix view
// ---------------------------------------------------------------------------
__global__ void rowmax_kernel(const float* __restrict__ X, float* __restrict__ R) {
  int idx = blockIdx.x * blockDim.x + threadIdx.x;
  if (idx >= 2 * NP) return;
  int b = idx / NP, p = idx % NP;
  const float* xp = X + (size_t)b * POOLN + (size_t)p * NP;
  float m = -3.4e38f;
  for (int q = 0; q < NP; ++q) m = fmaxf(m, xp[q]);
  R[idx] = m;
}

__global__ void colmax_kernel(const float* __restrict__ X, float* __restrict__ Cm) {
  int idx = blockIdx.x * blockDim.x + threadIdx.x;
  if (idx >= 2 * NP) return;
  int b = idx / NP, q = idx % NP;
  const float* xb = X + (size_t)b * POOLN;
  float m = -3.4e38f;
  for (int p = 0; p < NP; ++p) m = fmaxf(m, xb[(size_t)p * NP + q]);
  Cm[idx] = m;
}

__global__ void mm_apply_kernel(const float* __restrict__ X,
                                const float* __restrict__ R,
                                const float* __restrict__ Cm,
                                float* __restrict__ O) {
  int idx = blockIdx.x * blockDim.x + threadIdx.x;
  if (idx >= 2 * POOLN) return;
  int b = idx / POOLN, r = idx % POOLN;
  int p = r / NP, q = r % NP;
  float c = X[idx];
  float mA = R[b * NP + p] + 1e-5f;   // max over (d3,d4) for this (d1,d2)
  float mB = Cm[b * NP + q] + 1e-5f;  // max over (d1,d2) for this (d3,d4)
  O[idx] = c * (c / mA) * (c / mB);
}

// ---------------------------------------------------------------------------
// 5) Direct 3^4 SAME conv4d + ReLU on (2, Cin, 20,20,20,20)
// ---------------------------------------------------------------------------
__global__ void conv4d_relu_kernel(const float* __restrict__ in,
                                   const float* __restrict__ w,
                                   const float* __restrict__ bias,
                                   float* __restrict__ out,
                                   int Cin, int Cout) {
  int idx = blockIdx.x * blockDim.x + threadIdx.x;
  if (idx >= 2 * Cout * POOLN) return;
  int d4 = idx % 20; int t = idx / 20;
  int d3 = t % 20;   t /= 20;
  int d2 = t % 20;   t /= 20;
  int d1 = t % 20;   t /= 20;
  int co = t % Cout; int b = t / Cout;

  float acc = bias[co];
  for (int ci = 0; ci < Cin; ++ci) {
    const float* ip = in + (size_t)(b * Cin + ci) * POOLN;
    const float* wp = w + (size_t)(co * Cin + ci) * 81;
#pragma unroll
    for (int k1 = 0; k1 < 3; ++k1) {
      int x1 = d1 + k1 - 1; if ((unsigned)x1 >= 20u) continue;
#pragma unroll
      for (int k2 = 0; k2 < 3; ++k2) {
        int x2 = d2 + k2 - 1; if ((unsigned)x2 >= 20u) continue;
#pragma unroll
        for (int k3 = 0; k3 < 3; ++k3) {
          int x3 = d3 + k3 - 1; if ((unsigned)x3 >= 20u) continue;
#pragma unroll
          for (int k4 = 0; k4 < 3; ++k4) {
            int x4 = d4 + k4 - 1; if ((unsigned)x4 >= 20u) continue;
            acc += wp[k1 * 27 + k2 * 9 + k3 * 3 + k4] *
                   ip[x1 * 8000 + x2 * 400 + x3 * 20 + x4];
          }
        }
      }
    }
  }
  out[idx] = fmaxf(acc, 0.0f);
}

// ---------------------------------------------------------------------------
// 6) (d1,d2)<->(d3,d4) transpose = 400x400 matrix transpose per batch,
//    and fused add of the transposed second branch
// ---------------------------------------------------------------------------
__global__ void transpose_pq_kernel(const float* __restrict__ in,
                                    float* __restrict__ out) {
  int idx = blockIdx.x * blockDim.x + threadIdx.x;
  if (idx >= 2 * POOLN) return;
  int b = idx / POOLN, r = idx % POOLN;
  int p = r / NP, q = r % NP;
  out[(size_t)b * POOLN + (size_t)q * NP + p] = in[idx];
}

__global__ void add_transposed_kernel(const float* __restrict__ y1,
                                      const float* __restrict__ y2t,
                                      float* __restrict__ out) {
  int idx = blockIdx.x * blockDim.x + threadIdx.x;
  if (idx >= 2 * POOLN) return;
  int b = idx / POOLN, r = idx % POOLN;
  int p = r / NP, q = r % NP;
  out[idx] = y1[idx] + y2t[(size_t)b * POOLN + (size_t)q * NP + p];
}

// ---------------------------------------------------------------------------
// Host orchestration
// ---------------------------------------------------------------------------
extern "C" void kernel_launch(void* const* d_in, const int* in_sizes, int n_in,
                              void* d_out, int out_size, void* d_ws, size_t ws_size,
                              hipStream_t stream) {
  (void)in_sizes; (void)n_in; (void)out_size; (void)ws_size;
  const float* fA = (const float*)d_in[0];
  const float* fB = (const float*)d_in[1];
  const float* w1 = (const float*)d_in[2];
  const float* b1 = (const float*)d_in[3];
  const float* w2 = (const float*)d_in[4];
  const float* b2 = (const float*)d_in[5];
  const float* w3 = (const float*)d_in[6];
  const float* b3 = (const float*)d_in[7];
  float* out = (float*)d_out;

  char* ws = (char*)d_ws;
  size_t off = 0;
  auto take = [&](size_t bytes) -> char* {
    char* p = ws + off;
    off = (off + bytes + 255) & ~(size_t)255;
    return p;
  };
  _Float16* fA16 = (_Float16*)take(2ull * NPIX * NCH * 2);    //  6.55 MB
  _Float16* fB16 = (_Float16*)take(2ull * NPIX * NCH * 2);    //  6.55 MB
  float* corr   = (float*)take(2ull * CORR_B * 4);            // 20.48 MB
  float* pooled = (float*)take(2ull * POOLN * 4);
  float* Xmm    = (float*)take(2ull * POOLN * 4);             // conv input
  float* rm     = (float*)take(2ull * NP * 4);
  float* cm     = (float*)take(2ull * NP * 4);
  float* y1     = (float*)take(2ull * POOLN * 4);
  float* xT     = (float*)take(2ull * POOLN * 4);
  float* y2t    = (float*)take(2ull * POOLN * 4);
  float* ysum   = (float*)take(2ull * POOLN * 4);
  // Aliased 10-channel conv ping-pong buffers (12.8 MB each); the aliased
  // regions (f16 features / full corr) are dead by the time convs run.
  float* bufA = (float*)fA16;
  float* bufB = corr;

  dim3 blk(256);
  auto g = [](int n) { return dim3((unsigned)((n + 255) / 256)); };

  // 1) normalize + cast
  l2norm_f16_kernel<<<g(2 * NPIX), blk, 0, stream>>>(fA, fA16);
  l2norm_f16_kernel<<<g(2 * NPIX), blk, 0, stream>>>(fB, fB16);

  // 2) correlation GEMM (TDM -> LDS -> WMMA): block per (batch, M-tile)
  corr_wmma_kernel<<<dim3(200), blk, 0, stream>>>(fA16, fB16, corr);

  // 3) max-pool 2^4
  maxpool4d_kernel<<<g(2 * POOLN), blk, 0, stream>>>(corr, pooled);

  // 4) first mutual matching -> Xmm
  rowmax_kernel<<<g(2 * NP), blk, 0, stream>>>(pooled, rm);
  colmax_kernel<<<g(2 * NP), blk, 0, stream>>>(pooled, cm);
  mm_apply_kernel<<<g(2 * POOLN), blk, 0, stream>>>(pooled, rm, cm, Xmm);

  // 5) neigh_consensus branch 1: conv stack on Xmm -> y1
  conv4d_relu_kernel<<<g(2 * 10 * POOLN), blk, 0, stream>>>(Xmm, w1, b1, bufA, 1, 10);
  conv4d_relu_kernel<<<g(2 * 10 * POOLN), blk, 0, stream>>>(bufA, w2, b2, bufB, 10, 10);
  conv4d_relu_kernel<<<g(2 * 1 * POOLN), blk, 0, stream>>>(bufB, w3, b3, y1, 10, 1);

  //    branch 2: transpose -> conv stack -> (left transposed, fused on add)
  transpose_pq_kernel<<<g(2 * POOLN), blk, 0, stream>>>(Xmm, xT);
  conv4d_relu_kernel<<<g(2 * 10 * POOLN), blk, 0, stream>>>(xT, w1, b1, bufA, 1, 10);
  conv4d_relu_kernel<<<g(2 * 10 * POOLN), blk, 0, stream>>>(bufA, w2, b2, bufB, 10, 10);
  conv4d_relu_kernel<<<g(2 * 1 * POOLN), blk, 0, stream>>>(bufB, w3, b3, y2t, 10, 1);

  add_transposed_kernel<<<g(2 * POOLN), blk, 0, stream>>>(y1, y2t, ysum);

  // 6) final mutual matching straight into d_out
  rowmax_kernel<<<g(2 * NP), blk, 0, stream>>>(ysum, rm);
  colmax_kernel<<<g(2 * NP), blk, 0, stream>>>(ysum, cm);
  mm_apply_kernel<<<g(2 * POOLN), blk, 0, stream>>>(ysum, rm, cm, out);
}
